// TokenSelector_13907104105006
// MI455X (gfx1250) — compile-verified, matrix operand order checked by reading, not run on hardware
//
#include <hip/hip_runtime.h>
#include <hip/hip_bf16.h>

// ---------------- constants ----------------
#define B_   16
#define T_   4096
#define J_   17
#define C_   128
#define H_   512
#define K_   2048
#define JC_  (J_ * C_)          // 2176
#define BT_  (B_ * T_)          // 65536
#define BTC_ ((size_t)BT_ * C_) // 8388608

typedef __attribute__((ext_vector_type(16))) __bf16 v16bf;
typedef __attribute__((ext_vector_type(8)))  float  v8f;

// bf16 helpers (bit-level, avoids scalar __bf16 ABI issues on host pass)
static __device__ __forceinline__ unsigned short f2bf(float f) {
    unsigned u = __float_as_uint(f);
    u += 0x7FFFu + ((u >> 16) & 1u);   // round-to-nearest-even
    return (unsigned short)(u >> 16);
}
static __device__ __forceinline__ float bf2f(unsigned short h) {
    return __uint_as_float(((unsigned)h) << 16);
}

// ---------------- kernel 1: joint mean + LayerNorm -> st_bf (bf16) ----------
// one wave32 per (b,t) row; 4 channels per lane
__global__ __launch_bounds__(256)
void k_jointmean_ln(const float* __restrict__ tokens,
                    const float* __restrict__ ln_w,
                    const float* __restrict__ ln_b,
                    unsigned short* __restrict__ st_bf) {
    int lane = threadIdx.x & 31;
    int row  = (blockIdx.x * blockDim.x + threadIdx.x) >> 5;   // 0..BT_-1
    const float* base = tokens + (size_t)row * JC_ + lane * 4;
    float a0 = 0.f, a1 = 0.f, a2 = 0.f, a3 = 0.f;
#pragma unroll
    for (int j = 0; j < J_; ++j) {
        float4 v = *(const float4*)(base + j * C_);
        a0 += v.x; a1 += v.y; a2 += v.z; a3 += v.w;
    }
    const float invJ = 1.0f / (float)J_;
    a0 *= invJ; a1 *= invJ; a2 *= invJ; a3 *= invJ;

    float s = a0 + a1 + a2 + a3;
#pragma unroll
    for (int m = 1; m < 32; m <<= 1) s += __shfl_xor(s, m, 32);
    float mu = s * (1.0f / (float)C_);

    float q = a0 * a0 + a1 * a1 + a2 * a2 + a3 * a3;
#pragma unroll
    for (int m = 1; m < 32; m <<= 1) q += __shfl_xor(q, m, 32);
    float var = q * (1.0f / (float)C_) - mu * mu;
    float r = rsqrtf(var + 1e-5f);

    float4 w = *(const float4*)(ln_w + lane * 4);
    float4 b = *(const float4*)(ln_b + lane * 4);
    float y0 = (a0 - mu) * r * w.x + b.x;
    float y1 = (a1 - mu) * r * w.y + b.y;
    float y2 = (a2 - mu) * r * w.z + b.z;
    float y3 = (a3 - mu) * r * w.w + b.w;

    uint2 p;
    p.x = (unsigned)f2bf(y0) | ((unsigned)f2bf(y1) << 16);
    p.y = (unsigned)f2bf(y2) | ((unsigned)f2bf(y3) << 16);
    *(uint2*)(st_bf + (size_t)row * C_ + lane * 4) = p;
}

// ---------------- kernel 2: pack W1 (f32 [C,H]) into bf16 WMMA B fragments --
// layout: frag f = nt*4+ks; per lane 16 bf16 contiguous (two uint4 loads)
__global__ __launch_bounds__(256)
void k_pack_w1(const float* __restrict__ W1, unsigned short* __restrict__ w1p) {
    int idx  = blockIdx.x * 256 + threadIdx.x;   // 65536 total
    int i    = idx & 15;                         // element within lane fragment
    int lane = (idx >> 4) & 31;
    int f    = idx >> 9;                         // 0..127
    int ks   = f & 3;
    int nt   = f >> 2;
    int v    = i >> 1;
    int k    = ks * 32 + ((v < 4) ? 0 : 16) + ((lane >= 16) ? 8 : 0) + (v & 3) * 2 + (i & 1);
    int n    = nt * 16 + (lane & 15);
    w1p[(f * 32 + lane) * 16 + i] = f2bf(W1[k * H_ + n]);
}

// ---------------- kernel 2b: pack (b1, W2) pairs for one b64 load per tile --
__global__ __launch_bounds__(256)
void k_pack_bw(const float* __restrict__ b1, const float* __restrict__ W2,
               float2* __restrict__ bw2) {
    int n = blockIdx.x * 256 + threadIdx.x;      // 512 total
    float2 v; v.x = b1[n]; v.y = W2[n];
    bw2[n] = v;
}

// ---------------- kernel 3: gc[b][c] = mean_t st_n ---------------------------
__global__ __launch_bounds__(256)
void k_gc(const unsigned short* __restrict__ st_bf, float* __restrict__ gc) {
    int idx = blockIdx.x * 256 + threadIdx.x;    // 2048 = B*C
    int b = idx >> 7, c = idx & 127;
    const unsigned short* p = st_bf + (size_t)b * T_ * C_ + c;
    float s = 0.f;
    for (int t = 0; t < T_; ++t) s += bf2f(p[(size_t)t * C_]);
    gc[idx] = s * (1.0f / (float)T_);
}

// ---------------- kernel 4: st_bf += gc (in place) ---------------------------
__global__ __launch_bounds__(256)
void k_add_gc(unsigned short* __restrict__ st_bf, const float* __restrict__ gc) {
    size_t i = ((size_t)blockIdx.x * 256 + threadIdx.x) * 4;   // over BTC_
    int c = (int)(i & (C_ - 1));
    int b = (int)(i >> 19);                                    // T_*C_ = 2^19
    const float* g = gc + b * C_ + c;
    uint2 p = *(uint2*)(st_bf + i);
    float f0 = bf2f((unsigned short)(p.x & 0xFFFF)) + g[0];
    float f1 = bf2f((unsigned short)(p.x >> 16))    + g[1];
    float f2 = bf2f((unsigned short)(p.y & 0xFFFF)) + g[2];
    float f3 = bf2f((unsigned short)(p.y >> 16))    + g[3];
    p.x = (unsigned)f2bf(f0) | ((unsigned)f2bf(f1) << 16);
    p.y = (unsigned)f2bf(f2) | ((unsigned)f2bf(f3) << 16);
    *(uint2*)(st_bf + i) = p;
}

// branchless GELU (tanh form): h * sigmoid(1.59576919*(h + 0.044715 h^3))
// uses v_exp_f32 + v_rcp_f32; no exec-mask branches in the hot loop
static __device__ __forceinline__ float gelu_fast(float h) {
    float z = h * (1.5957691216057308f + 0.0713548162726009f * h * h);
    float e = __expf(-z);
    return h * __builtin_amdgcn_rcpf(1.0f + e);
}

// ---------------- kernel 5: WMMA scoring GEMM + fused GELU + W2 dot ----------
// one wave per 16 rows; scores[row] = gelu((x @ W1 + b1)) @ W2 + b2
__global__ __launch_bounds__(256)
void k_score_gemm(const unsigned short* __restrict__ x_bf,
                  const unsigned short* __restrict__ w1p,
                  const float2* __restrict__ bw2,
                  const float* __restrict__ b2,
                  float* __restrict__ scores) {
    int lane = threadIdx.x & 31;
    int wave = blockIdx.x * (blockDim.x >> 5) + (threadIdx.x >> 5);
    int rowBase = wave * 16;

    // A fragments: 16x32 bf16 per k-step; lane = row (M), two 16-byte runs
    const unsigned short* Arow = x_bf + (size_t)(rowBase + (lane & 15)) * C_;
    int kboff = (lane >> 4) * 8;
    union AU { uint4 q[2]; v16bf v; };
    v16bf a[4];
#pragma unroll
    for (int ks = 0; ks < 4; ++ks) {
        AU u;
        int kb = ks * 32 + kboff;
        u.q[0] = *(const uint4*)(Arow + kb);
        u.q[1] = *(const uint4*)(Arow + kb + 16);
        a[ks] = u.v;
    }

    v8f acc = {0.f, 0.f, 0.f, 0.f, 0.f, 0.f, 0.f, 0.f};
    const uint4* Bp = (const uint4*)w1p;

#pragma unroll 2
    for (int nt = 0; nt < 32; ++nt) {
        v8f c = {0.f, 0.f, 0.f, 0.f, 0.f, 0.f, 0.f, 0.f};
#pragma unroll
        for (int ks = 0; ks < 4; ++ks) {
            AU ub;
            int f = nt * 4 + ks;
            ub.q[0] = Bp[(f * 32 + lane) * 2 + 0];
            ub.q[1] = Bp[(f * 32 + lane) * 2 + 1];
            c = __builtin_amdgcn_wmma_f32_16x16x32_bf16(
                    false, a[ks], false, ub.v, (short)0, c, false, false);
        }
        int n = nt * 16 + (lane & 15);
        float2 bw = bw2[n];           // {b1[n], W2[n]} one b64 load
#pragma unroll
        for (int g = 0; g < 8; ++g) {
            float h = gelu_fast(c[g] + bw.x);
            acc[g] += h * bw.y;
        }
    }
    // reduce over the 16 lanes sharing each row-half
#pragma unroll
    for (int m = 1; m < 16; m <<= 1)
#pragma unroll
        for (int g = 0; g < 8; ++g) acc[g] += __shfl_xor(acc[g], m, 32);

    float bb = b2[0];
    if (lane == 0) {
#pragma unroll
        for (int g = 0; g < 8; ++g) scores[rowBase + g] = acc[g] + bb;        // M=g
    } else if (lane == 16) {
#pragma unroll
        for (int g = 0; g < 8; ++g) scores[rowBase + 8 + g] = acc[g] + bb;    // M=g+8
    }
}

// ---------------- kernel 6: per-batch score standardization ------------------
__global__ __launch_bounds__(256)
void k_standardize(float* __restrict__ scores, float* __restrict__ out_scores) {
    int b = blockIdx.x, tid = threadIdx.x;
    __shared__ float ssum[256], ssq[256];
    float s = 0.f, q = 0.f;
    for (int t = tid; t < T_; t += 256) {
        float v = scores[b * T_ + t];
        s += v; q += v * v;
    }
    ssum[tid] = s; ssq[tid] = q;
    __syncthreads();
    for (int st = 128; st > 0; st >>= 1) {
        if (tid < st) { ssum[tid] += ssum[tid + st]; ssq[tid] += ssq[tid + st]; }
        __syncthreads();
    }
    float mean = ssum[0] * (1.0f / (float)T_);
    float var  = ssq[0] * (1.0f / (float)T_) - mean * mean;
    float inv  = 1.0f / (sqrtf(fmaxf(var, 0.f)) + 1e-6f);
    for (int t = tid; t < T_; t += 256) {
        float v = (scores[b * T_ + t] - mean) * inv;
        scores[b * T_ + t] = v;
        out_scores[b * T_ + t] = v;
    }
}

// ---------------- kernel 7: per-batch top-K via bitonic sort in LDS ----------
__global__ __launch_bounds__(1024)
void k_topk(const float* __restrict__ scores,
            int* __restrict__ kept_idx, float* __restrict__ kept_score,
            float* __restrict__ out_idx_f) {
    int b = blockIdx.x, tid = threadIdx.x;
    __shared__ float skey[T_];
    __shared__ int   sidx[T_];
    for (int i = tid; i < T_; i += 1024) { skey[i] = scores[b * T_ + i]; sidx[i] = i; }
    __syncthreads();

    // bitonic sort descending by score
    for (int k2 = 2; k2 <= T_; k2 <<= 1) {
        for (int j = k2 >> 1; j > 0; j >>= 1) {
            for (int p = tid; p < T_ / 2; p += 1024) {
                int i   = 2 * p - (p & (j - 1));
                int ixj = i | j;
                bool up = ((i & k2) == 0);
                float av = skey[i], cv = skey[ixj];
                bool sw = up ? (av < cv) : (av > cv);
                if (sw) {
                    skey[i] = cv; skey[ixj] = av;
                    int t = sidx[i]; sidx[i] = sidx[ixj]; sidx[ixj] = t;
                }
            }
            __syncthreads();
        }
    }
    // re-sort kept K_ entries by index ascending (carry score)
    for (int k2 = 2; k2 <= K_; k2 <<= 1) {
        for (int j = k2 >> 1; j > 0; j >>= 1) {
            for (int p = tid; p < K_ / 2; p += 1024) {
                int i   = 2 * p - (p & (j - 1));
                int ixj = i | j;
                bool up = ((i & k2) == 0);
                int ai = sidx[i], ci = sidx[ixj];
                bool sw = up ? (ai > ci) : (ai < ci);
                if (sw) {
                    sidx[i] = ci; sidx[ixj] = ai;
                    float t = skey[i]; skey[i] = skey[ixj]; skey[ixj] = t;
                }
            }
            __syncthreads();
        }
    }
    for (int i = tid; i < K_; i += 1024) {
        kept_idx[b * K_ + i]   = sidx[i];
        kept_score[b * K_ + i] = skey[i];
        out_idx_f[b * K_ + i]  = (float)sidx[i];
    }
}

// ---------------- kernel 8: gather + sigmoid gate ----------------------------
__global__ __launch_bounds__(256)
void k_gather(const float* __restrict__ tokens,
              const int* __restrict__ kept_idx,
              const float* __restrict__ kept_score,
              float* __restrict__ out_sel) {
    int row = blockIdx.x;                 // b*K_ + k
    int b   = row >> 11;                  // K_ = 2048
    int idx = kept_idx[row];
    float sc = kept_score[row];
    float g = 1.0f + 0.1f / (1.0f + __expf(-sc));
    const float4* src = (const float4*)(tokens + ((size_t)b * T_ + idx) * JC_);
    float4*       dst = (float4*)(out_sel + (size_t)row * JC_);
    for (int i = threadIdx.x; i < JC_ / 4; i += 256) {
        float4 v = src[i];
        v.x *= g; v.y *= g; v.z *= g; v.w *= g;
        dst[i] = v;
    }
}

// ---------------- launcher ---------------------------------------------------
extern "C" void kernel_launch(void* const* d_in, const int* in_sizes, int n_in,
                              void* d_out, int out_size, void* d_ws, size_t ws_size,
                              hipStream_t stream) {
    const float* tokens = (const float*)d_in[0];
    const float* ln_w   = (const float*)d_in[1];
    const float* ln_b   = (const float*)d_in[2];
    const float* W1     = (const float*)d_in[3];
    const float* b1     = (const float*)d_in[4];
    const float* W2     = (const float*)d_in[5];
    const float* b2     = (const float*)d_in[6];

    char* ws = (char*)d_ws;
    unsigned short* st_bf    = (unsigned short*)(ws);                 // 16 MB
    float*          gc       = (float*)(ws + 16777216);               // 8 KB
    unsigned short* w1p      = (unsigned short*)(ws + 16785408);      // 128 KB
    float*          scoresb  = (float*)(ws + 16916480);               // 256 KB
    int*            kept_idx = (int*)(ws + 17178624);                 // 128 KB
    float*          kept_sc  = (float*)(ws + 17309696);               // 128 KB
    float2*         bw2      = (float2*)(ws + 17440768);              // 4 KB

    float* out        = (float*)d_out;
    float* out_sel    = out;                                          // B*K*J*C
    float* out_idx    = out + (size_t)B_ * K_ * JC_;                  // B*K
    float* out_scores = out_idx + (size_t)B_ * K_;                    // B*T

    k_jointmean_ln<<<BT_ / 8, 256, 0, stream>>>(tokens, ln_w, ln_b, st_bf);
    k_pack_w1<<<256, 256, 0, stream>>>(W1, w1p);
    k_pack_bw<<<2, 256, 0, stream>>>(b1, W2, bw2);
    k_gc<<<(B_ * C_) / 256, 256, 0, stream>>>(st_bf, gc);
    k_add_gc<<<(int)(BTC_ / 4 / 256), 256, 0, stream>>>(st_bf, gc);
    k_score_gemm<<<BT_ / 16 / 8, 256, 0, stream>>>(st_bf, w1p, bw2, b2, scoresb);
    k_standardize<<<B_, 256, 0, stream>>>(scoresb, out_scores);
    k_topk<<<B_, 1024, 0, stream>>>(scoresb, kept_idx, kept_sc, out_idx);
    k_gather<<<B_ * K_, 256, 0, stream>>>(tokens, kept_idx, kept_sc, out_sel);
}
